// PhiMoESparseMoeBlock_80083960201403
// MI455X (gfx1250) — matrix operand order repaired and checked by reading, not compiled, not run on hardware
//
#include <hip/hip_runtime.h>
#include <cstdint>
#include <cstddef>

// Problem constants (from reference)
#define S_TOK 2048
#define H_DIM 2048
#define I_DIM 5632
#define N_EXP 8
#define JITTER 0.01f
#define LDS_PAD 40   // row stride (bf16 elems) for 32-wide K tiles: 80B -> conflict-free b128 reads

typedef __attribute__((ext_vector_type(16))) __bf16 v16bf;
typedef __attribute__((ext_vector_type(8)))  float v8f;

union Frag {
    v16bf v;
    unsigned short u[16];
    uint4 q[2];
};

__device__ __forceinline__ unsigned short f2bf(float f) {
    unsigned int b = __float_as_uint(f);
    unsigned int r = b + 0x7FFFu + ((b >> 16) & 1u); // RNE (used only in small epilogues)
    return (unsigned short)(r >> 16);
}

// Pack two fp32 -> one dword of two bf16 (round-half-up) : 2x v_add + 1x v_perm_b32
__device__ __forceinline__ unsigned int pk2bf(float lo, float hi) {
    unsigned int bl = __float_as_uint(lo) + 0x8000u;
    unsigned int bh = __float_as_uint(hi) + 0x8000u;
    // result bytes = [bl.b2, bl.b3, bh.b2, bh.b3]
    return __builtin_amdgcn_perm(bh, bl, 0x07060302u);
}

// Convert 16 contiguous fp32 -> 16 bf16 and store to LDS (two b128 stores)
__device__ __forceinline__ void fill16(const float* __restrict__ src, unsigned short* dst) {
    float4 c0 = *(const float4*)(src);
    float4 c1 = *(const float4*)(src + 4);
    float4 c2 = *(const float4*)(src + 8);
    float4 c3 = *(const float4*)(src + 12);
    uint4 q0, q1;
    q0.x = pk2bf(c0.x, c0.y); q0.y = pk2bf(c0.z, c0.w);
    q0.z = pk2bf(c1.x, c1.y); q0.w = pk2bf(c1.z, c1.w);
    q1.x = pk2bf(c2.x, c2.y); q1.y = pk2bf(c2.z, c2.w);
    q1.z = pk2bf(c3.x, c3.y); q1.w = pk2bf(c3.z, c3.w);
    *(uint4*)dst = q0;
    *((uint4*)dst + 1) = q1;
}

// A-fragment (16x32): lane group g holds K chunks [8g,8g+8) and [16+8g,16+8g+8)
__device__ __forceinline__ v16bf lds_frag_a(const unsigned short* As, int row, int g) {
    Frag fr;
    const unsigned short* p = As + row * LDS_PAD + g * 8;
    fr.q[0] = *(const uint4*)(p);
    fr.q[1] = *(const uint4*)(p + 16);
    return fr.v;
}

// B-fragment (32x16): lane holds row n=lane&15; group g holds K chunks [8g,8g+8) and [16+8g,16+8g+8)
__device__ __forceinline__ v16bf lds_frag_b(const unsigned short* Bs, int row, int g) {
    Frag fr;
    const unsigned short* p = Bs + row * LDS_PAD + g * 8;
    fr.q[0] = *(const uint4*)(p);
    fr.q[1] = *(const uint4*)(p + 16);
    return fr.v;
}

// A-fragment directly from bf16 global memory (hid buffer)
__device__ __forceinline__ v16bf load_a_bf16(const __bf16* __restrict__ row, int kk, int g) {
    Frag fr;
    const int k0 = kk + g * 8;
    fr.q[0] = *(const uint4*)(row + k0);
    fr.q[1] = *(const uint4*)(row + k0 + 16);
    return fr.v;
}

// ---------------------------------------------------------------------------
// Kernel 1: gating logits + sparsemixer top-2 routing
// ---------------------------------------------------------------------------
__global__ __launch_bounds__(256) void moe_route_kernel(
    const float* __restrict__ x, const float* __restrict__ gw,
    int* __restrict__ cnt, int* __restrict__ ltok, float* __restrict__ lw) {
    __shared__ float sg[N_EXP * H_DIM];  // 64 KB of the 320 KB WGP LDS
    for (int i = threadIdx.x; i < N_EXP * H_DIM; i += blockDim.x) sg[i] = gw[i];
    __syncthreads();

    const int t = blockIdx.x * blockDim.x + threadIdx.x;
    if (t >= S_TOK) return;
    const float* xr = x + (size_t)t * H_DIM;

    float acc[N_EXP];
#pragma unroll
    for (int e = 0; e < N_EXP; ++e) acc[e] = 0.0f;
    for (int h = 0; h < H_DIM; h += 4) {
        float4 xv = *(const float4*)(xr + h);
#pragma unroll
        for (int e = 0; e < N_EXP; ++e) {
            const float* g = sg + e * H_DIM + h;
            acc[e] += xv.x * g[0] + xv.y * g[1] + xv.z * g[2] + xv.w * g[3];
        }
    }

    int i0 = 0; float m1 = acc[0];
#pragma unroll
    for (int e = 1; e < N_EXP; ++e) if (acc[e] > m1) { m1 = acc[e]; i0 = e; }
    int i1 = (i0 == 0) ? 1 : 0; float m2 = acc[i1];
#pragma unroll
    for (int e = 0; e < N_EXP; ++e)
        if (e != i0 && acc[e] > m2) { m2 = acc[e]; i1 = e; }

    float den1 = 0.0f;
#pragma unroll
    for (int e = 0; e < N_EXP; ++e) {
        bool masked = ((m1 - acc[e]) / fmaxf(fabsf(acc[e]), m1)) > 2.0f * JITTER;
        den1 += masked ? 0.0f : __expf(acc[e] - m1);
    }
    float mult1 = 1.0f / den1;

    float den2 = 0.0f;
#pragma unroll
    for (int e = 0; e < N_EXP; ++e) {
        bool masked = (e == i0) ||
                      (((m2 - acc[e]) / fmaxf(fabsf(acc[e]), m2)) > 2.0f * JITTER);
        den2 += masked ? 0.0f : __expf(acc[e] - m2);
    }
    float mult2 = 1.0f / den2;

    int s0 = atomicAdd(&cnt[i0], 1);
    ltok[i0 * S_TOK + s0] = t;
    lw[i0 * S_TOK + s0] = mult1;
    int s1 = atomicAdd(&cnt[i1], 1);
    ltok[i1 * S_TOK + s1] = t;
    lw[i1 * S_TOK + s1] = mult2;
}

// ---------------------------------------------------------------------------
// Kernel 2: exclusive prefix sum of expert counts
// ---------------------------------------------------------------------------
__global__ void moe_prefix_kernel(const int* __restrict__ cnt, int* __restrict__ offs) {
    if (threadIdx.x == 0 && blockIdx.x == 0) {
        int a = 0;
#pragma unroll
        for (int e = 0; e < N_EXP; ++e) { offs[e] = a; a += cnt[e]; }
    }
}

// ---------------------------------------------------------------------------
// Kernel 3: FFN1  hid = silu(x @ w1^T) * (x @ w3^T)
// Block 256 thr = 8 waves; block tile 128M x 128N; K-step 32.
// A/B tiles staged in LDS as bf16 (converted once per block).
// ---------------------------------------------------------------------------
__global__ __launch_bounds__(256) void moe_ffn1_kernel(
    const float* __restrict__ x, const float* __restrict__ w1,
    const float* __restrict__ w3, const int* __restrict__ cnt,
    const int* __restrict__ offs, const int* __restrict__ ltok,
    unsigned short* __restrict__ hid) {
    __shared__ unsigned short As[128 * LDS_PAD];
    __shared__ unsigned short B1s[128 * LDS_PAD];
    __shared__ unsigned short B3s[128 * LDS_PAD];

    const int e = blockIdx.z;
    const int count = cnt[e];
    const int m0 = blockIdx.y * 128;
    if (m0 >= count) return;
    const int n0 = blockIdx.x * 128;

    const int tid = threadIdx.x;
    const int wave = tid >> 5;
    const int lane = tid & 31;
    const int g = lane >> 4;
    const int ln = lane & 15;
    const int wm = wave >> 1;       // 0..3 -> 32-row wave tile
    const int wn = wave & 1;        // 0..1 -> 64-col wave tile

    // Fill assignment: each thread owns one half-row (16 K values) of each tile
    const int frow = tid >> 1;      // 0..127
    const int fk   = (tid & 1) * 16;
    const int mfill = m0 + frow;
    const int mc = (mfill < count) ? mfill : (count - 1);
    const float* arow = x + (size_t)ltok[e * S_TOK + mc] * H_DIM + fk;
    const float* b1row = w1 + (size_t)e * I_DIM * H_DIM + (size_t)(n0 + frow) * H_DIM + fk;
    const float* b3row = w3 + (size_t)e * I_DIM * H_DIM + (size_t)(n0 + frow) * H_DIM + fk;
    unsigned short* adst  = As  + frow * LDS_PAD + fk;
    unsigned short* b1dst = B1s + frow * LDS_PAD + fk;
    unsigned short* b3dst = B3s + frow * LDS_PAD + fk;

    const v8f zacc = {0.f, 0.f, 0.f, 0.f, 0.f, 0.f, 0.f, 0.f};
    v8f accG[2][4], accU[2][4];
#pragma unroll
    for (int f = 0; f < 2; ++f)
#pragma unroll
        for (int j = 0; j < 4; ++j) { accG[f][j] = zacc; accU[f][j] = zacc; }

    for (int kk = 0; kk < H_DIM; kk += 32) {
        fill16(arow + kk, adst);
        fill16(b1row + kk, b1dst);
        fill16(b3row + kk, b3dst);
        __builtin_prefetch(b1row + kk + 64, 0, 1);   // global_prefetch_b8
        __builtin_prefetch(b3row + kk + 64, 0, 1);
        __syncthreads();

        v16bf a[2];
#pragma unroll
        for (int f = 0; f < 2; ++f) a[f] = lds_frag_a(As, wm * 32 + f * 16 + ln, g);
#pragma unroll
        for (int j = 0; j < 4; ++j) {
            const int brow = wn * 64 + j * 16 + ln;
            v16bf b1f = lds_frag_b(B1s, brow, g);
            v16bf b3f = lds_frag_b(B3s, brow, g);
#pragma unroll
            for (int f = 0; f < 2; ++f) {
                accG[f][j] = __builtin_amdgcn_wmma_f32_16x16x32_bf16(
                    false, a[f], false, b1f, (short)0, accG[f][j], false, false);
                accU[f][j] = __builtin_amdgcn_wmma_f32_16x16x32_bf16(
                    false, a[f], false, b3f, (short)0, accU[f][j], false, false);
            }
        }
        __syncthreads();
    }

    const int rowoff = offs[e];
    const int mbase = m0 + wm * 32;
    const int nbase = n0 + wn * 64;
#pragma unroll
    for (int f = 0; f < 2; ++f)
#pragma unroll
        for (int j = 0; j < 4; ++j) {
            const int n = nbase + j * 16 + ln;
#pragma unroll
            for (int r = 0; r < 8; ++r) {
                int m = mbase + f * 16 + g * 8 + r;  // C layout: lanes16-31 -> M+8
                if (m < count) {
                    float gv = accG[f][j][r];
                    float uv = accU[f][j][r];
                    float hv = gv * (1.0f / (1.0f + __expf(-gv))) * uv;  // silu * up
                    hid[(size_t)(rowoff + m) * I_DIM + n] = f2bf(hv);
                }
            }
        }
}

// ---------------------------------------------------------------------------
// Kernel 4: FFN2  out[tok] += w * (hid @ w2^T)
// A read directly from bf16 hid; B (w2, fp32) staged in LDS as bf16.
// ---------------------------------------------------------------------------
__global__ __launch_bounds__(256) void moe_ffn2_kernel(
    const unsigned short* __restrict__ hid, const float* __restrict__ w2,
    const int* __restrict__ cnt, const int* __restrict__ offs,
    const int* __restrict__ ltok, const float* __restrict__ lw,
    float* __restrict__ out) {
    __shared__ unsigned short Bs[128 * LDS_PAD];

    const int e = blockIdx.z;
    const int count = cnt[e];
    const int m0 = blockIdx.y * 128;
    if (m0 >= count) return;
    const int n0 = blockIdx.x * 128;

    const int tid = threadIdx.x;
    const int wave = tid >> 5;
    const int lane = tid & 31;
    const int g = lane >> 4;
    const int ln = lane & 15;
    const int wm = wave >> 1;
    const int wn = wave & 1;
    const int rowoff = offs[e];
    const int mbase = m0 + wm * 32;
    const int nbase = n0 + wn * 64;

    const __bf16* hrow[2];
#pragma unroll
    for (int f = 0; f < 2; ++f) {
        int m = mbase + f * 16 + ln;
        int mc = (m < count) ? m : (count - 1);
        hrow[f] = (const __bf16*)hid + (size_t)(rowoff + mc) * I_DIM;
    }

    const int frow = tid >> 1;
    const int fk   = (tid & 1) * 16;
    const float* brow = w2 + (size_t)e * H_DIM * I_DIM + (size_t)(n0 + frow) * I_DIM + fk;
    unsigned short* bdst = Bs + frow * LDS_PAD + fk;

    const v8f zacc = {0.f, 0.f, 0.f, 0.f, 0.f, 0.f, 0.f, 0.f};
    v8f acc[2][4];
#pragma unroll
    for (int f = 0; f < 2; ++f)
#pragma unroll
        for (int j = 0; j < 4; ++j) acc[f][j] = zacc;

    for (int kk = 0; kk < I_DIM; kk += 32) {
        fill16(brow + kk, bdst);
        __builtin_prefetch(brow + kk + 64, 0, 1);   // global_prefetch_b8
        __syncthreads();

        v16bf a[2];
#pragma unroll
        for (int f = 0; f < 2; ++f) a[f] = load_a_bf16(hrow[f], kk, g);
#pragma unroll
        for (int j = 0; j < 4; ++j) {
            v16bf bf = lds_frag_b(Bs, wn * 64 + j * 16 + ln, g);
#pragma unroll
            for (int f = 0; f < 2; ++f)
                acc[f][j] = __builtin_amdgcn_wmma_f32_16x16x32_bf16(
                    false, a[f], false, bf, (short)0, acc[f][j], false, false);
        }
        __syncthreads();
    }

#pragma unroll
    for (int f = 0; f < 2; ++f)
#pragma unroll
        for (int r = 0; r < 8; ++r) {
            int m = mbase + f * 16 + g * 8 + r;
            if (m < count) {
                int tok = ltok[e * S_TOK + m];
                float w = lw[e * S_TOK + m];
#pragma unroll
                for (int j = 0; j < 4; ++j) {
                    int n = nbase + j * 16 + ln;
                    atomicAdd(&out[(size_t)tok * H_DIM + n], w * acc[f][j][r]);
                }
            }
        }
}

// ---------------------------------------------------------------------------
// Host launch
// ---------------------------------------------------------------------------
extern "C" void kernel_launch(void* const* d_in, const int* in_sizes, int n_in,
                              void* d_out, int out_size, void* d_ws, size_t ws_size,
                              hipStream_t stream) {
    const float* x  = (const float*)d_in[0];  // (1, 2048, 2048)
    const float* gw = (const float*)d_in[1];  // (8, 2048)
    const float* w1 = (const float*)d_in[2];  // (8, 5632, 2048)
    const float* w2 = (const float*)d_in[3];  // (8, 2048, 5632)
    const float* w3 = (const float*)d_in[4];  // (8, 5632, 2048)
    float* out = (float*)d_out;

    // Workspace layout
    char* ws = (char*)d_ws;
    int*   cnt  = (int*)ws;                                // 8 ints
    int*   offs = cnt + 8;                                 // 8 ints
    int*   ltok = offs + 8;                                // E*S ints
    float* lw   = (float*)(ltok + N_EXP * S_TOK);          // E*S floats
    size_t hid_off = ((size_t)((char*)(lw + N_EXP * S_TOK) - ws) + 255) & ~(size_t)255;
    unsigned short* hid = (unsigned short*)(ws + hid_off); // 4096 x 5632 bf16 (~46 MB)

    hipMemsetAsync(cnt, 0, 8 * sizeof(int), stream);
    hipMemsetAsync(out, 0, (size_t)S_TOK * H_DIM * sizeof(float), stream);

    moe_route_kernel<<<S_TOK / 256, 256, 0, stream>>>(x, gw, cnt, ltok, lw);
    moe_prefix_kernel<<<1, 32, 0, stream>>>(cnt, offs);

    dim3 g1(I_DIM / 128, S_TOK / 128, N_EXP);  // (44, 16, 8)
    moe_ffn1_kernel<<<g1, 256, 0, stream>>>(x, w1, w3, cnt, offs, ltok, hid);

    dim3 g2(H_DIM / 128, S_TOK / 128, N_EXP);  // (16, 16, 8)
    moe_ffn2_kernel<<<g2, 256, 0, stream>>>(hid, w2, cnt, offs, ltok, lw, out);
}
